// HybridBinaryClassifier360_65481071401114
// MI455X (gfx1250) — matrix operand order repaired and checked by reading, not compiled, not run on hardware
//
#include <hip/hip_runtime.h>
#include <hip/hip_bf16.h>
#include <math.h>

// ---------------------------------------------------------------------------
// Types for CDNA5 WMMA (wave32)
// ---------------------------------------------------------------------------
typedef __attribute__((ext_vector_type(16))) __bf16 v16bf;
typedef __attribute__((ext_vector_type(8)))  __bf16 v8bf;
typedef __attribute__((ext_vector_type(8)))  float  v8f;

// Padded GEMM dimensions
#define K1   55815            // fc1 reduction (15*61*61)
#define K1P  55840            // K1 padded to multiple of 32
#define N1P  128              // fc1 output cols padded (120 -> 128)
#define K2P  128              // fc2 reduction padded (120 -> 128)
#define N2P  96               // fc2 output cols padded (84 -> 96)

// ---------------------------------------------------------------------------
// conv1 (3->6, 5x5, stride 2, pad 1) + ReLU + 2x2/s1 maxpool, fused.
// x: [128,3,250,250] -> out: [128,6,123,123] fp32
// ---------------------------------------------------------------------------
__global__ void k_conv1pool(const float* __restrict__ x,
                            const float* __restrict__ w,
                            const float* __restrict__ bias,
                            float* __restrict__ out)
{
    const int P = 123;
    int pix = blockIdx.x * blockDim.x + threadIdx.x;
    if (pix >= P * P) return;
    int c = blockIdx.y;                // 0..5 (uniform per block)
    int b = blockIdx.z;                // 0..127
    int py = pix / P, px = pix % P;

    const float* xb = x + (size_t)b * 3 * 250 * 250;
    const float* wc = w + c * 3 * 5 * 5;

    float acc00 = 0.f, acc01 = 0.f, acc10 = 0.f, acc11 = 0.f;
    for (int ci = 0; ci < 3; ++ci) {
        const float* xc = xb + (size_t)ci * 250 * 250;
        const float* wk = wc + ci * 25;
#pragma unroll
        for (int ky = 0; ky < 5; ++ky) {
#pragma unroll
            for (int kx = 0; kx < 5; ++kx) {
                float wv = wk[ky * 5 + kx];
                int iy0 = py * 2 - 1 + ky;
                int iy1 = iy0 + 2;
                int ix0 = px * 2 - 1 + kx;
                int ix1 = ix0 + 2;
                bool y0 = (iy0 >= 0) & (iy0 < 250);
                bool y1 = (iy1 >= 0) & (iy1 < 250);
                bool x0 = (ix0 >= 0) & (ix0 < 250);
                bool x1 = (ix1 < 250);
                float v00 = (y0 & x0) ? xc[iy0 * 250 + ix0] : 0.f;
                float v01 = (y0 & x1) ? xc[iy0 * 250 + ix1] : 0.f;
                float v10 = (y1 & x0) ? xc[iy1 * 250 + ix0] : 0.f;
                float v11 = (y1 & x1) ? xc[iy1 * 250 + ix1] : 0.f;
                acc00 = fmaf(wv, v00, acc00);
                acc01 = fmaf(wv, v01, acc01);
                acc10 = fmaf(wv, v10, acc10);
                acc11 = fmaf(wv, v11, acc11);
            }
        }
    }
    float m = fmaxf(fmaxf(acc00, acc01), fmaxf(acc10, acc11)) + bias[c];
    m = fmaxf(m, 0.f);
    out[(((size_t)b * 6 + c) * P + py) * P + px] = m;
}

// ---------------------------------------------------------------------------
// conv2 (6->15, 3x3, stride 2, pad 1) + ReLU + 2x2/s1 maxpool, fused.
// in: [128,6,123,123] fp32 -> act: [128, K1P] bf16 (flattened, K-padded)
// ---------------------------------------------------------------------------
__global__ void k_conv2pool(const float* __restrict__ in,
                            const float* __restrict__ w,
                            const float* __restrict__ bias,
                            __bf16* __restrict__ act)
{
    const int S = 123;
    const int P = 61;
    int pix = blockIdx.x * blockDim.x + threadIdx.x;
    if (pix >= P * P) return;
    int c = blockIdx.y;                // 0..14
    int b = blockIdx.z;                // 0..127
    int py = pix / P, px = pix % P;

    const float* xb = in + (size_t)b * 6 * S * S;
    const float* wc = w + c * 6 * 3 * 3;

    float acc00 = 0.f, acc01 = 0.f, acc10 = 0.f, acc11 = 0.f;
    for (int ci = 0; ci < 6; ++ci) {
        const float* xc = xb + (size_t)ci * S * S;
        const float* wk = wc + ci * 9;
#pragma unroll
        for (int ky = 0; ky < 3; ++ky) {
#pragma unroll
            for (int kx = 0; kx < 3; ++kx) {
                float wv = wk[ky * 3 + kx];
                int iy0 = py * 2 - 1 + ky;
                int iy1 = iy0 + 2;
                int ix0 = px * 2 - 1 + kx;
                int ix1 = ix0 + 2;
                bool y0 = (iy0 >= 0) & (iy0 < S);
                bool y1 = (iy1 >= 0) & (iy1 < S);
                bool x0 = (ix0 >= 0) & (ix0 < S);
                bool x1 = (ix1 < S);
                float v00 = (y0 & x0) ? xc[iy0 * S + ix0] : 0.f;
                float v01 = (y0 & x1) ? xc[iy0 * S + ix1] : 0.f;
                float v10 = (y1 & x0) ? xc[iy1 * S + ix0] : 0.f;
                float v11 = (y1 & x1) ? xc[iy1 * S + ix1] : 0.f;
                acc00 = fmaf(wv, v00, acc00);
                acc01 = fmaf(wv, v01, acc01);
                acc10 = fmaf(wv, v10, acc10);
                acc11 = fmaf(wv, v11, acc11);
            }
        }
    }
    float m = fmaxf(fmaxf(acc00, acc01), fmaxf(acc10, acc11)) + bias[c];
    m = fmaxf(m, 0.f);
    act[(size_t)b * K1P + (size_t)c * (P * P) + py * P + px] = (__bf16)m;
}

// ---------------------------------------------------------------------------
// fp32 [N,K] -> bf16 [Np,Kp] with zero padding (one-shot weight conversion)
// ---------------------------------------------------------------------------
__global__ void k_cvt_pad_bf16(const float* __restrict__ src, int N, int K,
                               __bf16* __restrict__ dst, int Np, int Kp)
{
    size_t i = (size_t)blockIdx.x * blockDim.x + threadIdx.x;
    size_t total = (size_t)Np * Kp;
    if (i >= total) return;
    int n = (int)(i / Kp);
    int k = (int)(i % Kp);
    float v = (n < N && k < K) ? src[(size_t)n * K + k] : 0.f;
    dst[i] = (__bf16)v;
}

// ---------------------------------------------------------------------------
// Zero the K-pad tail of the activation matrix: act[b][K1..K1P) = 0
// ---------------------------------------------------------------------------
__global__ void k_zero_act_tail(__bf16* __restrict__ act)
{
    const int pad = K1P - K1;          // 25
    int i = blockIdx.x * blockDim.x + threadIdx.x;
    if (i >= 128 * pad) return;
    int b = i / pad;
    int k = K1 + (i % pad);
    act[(size_t)b * K1P + k] = (__bf16)0.f;
}

// ---------------------------------------------------------------------------
// WMMA GEMM: out[M,Np] = relu(A[M,Kp](bf16) * B[Np,Kp](bf16)^T + bias)
// One wave32 per 16x16 tile. Operands pre-padded: the K-loop is branch-free
// (two v8bf (b128) loads per fragment, one v_wmma per 32-K step).
//   outb != nullptr : store bf16 into [M,Np] (pad cols written as 0)
//   outb == nullptr : store fp32 into [M,Nvalid] with n < Nvalid guard
// grid = (M/16, Np/16), block = 32.
// ---------------------------------------------------------------------------
__global__ void k_fc_wmma(const __bf16* __restrict__ A, int lda,
                          const __bf16* __restrict__ B, int ldb,
                          const float* __restrict__ bias,
                          int Kp, int Nvalid,
                          __bf16* __restrict__ outb, int ldob,
                          float*  __restrict__ outf, int ldof)
{
    int m0   = blockIdx.x * 16;
    int n0   = blockIdx.y * 16;
    int lane = threadIdx.x;            // 0..31
    int half = lane >> 4;              // 0 or 1
    int l16  = lane & 15;
    int n    = n0 + l16;

    // A: lane holds row M = m0+l16; elems 0..7 -> K = k0 + half*8 + e,
    //                               elems 8..15 -> K = k0 + 16 + half*8 + e
    const __bf16* arow = A + (size_t)(m0 + l16) * lda;
    // B: lane holds col N = n; same K mapping as A
    const __bf16* brow = B + (size_t)n * ldb;

    v8f c = {};
    for (int k0 = 0; k0 < Kp; k0 += 32) {
        v8bf a0 = *(const v8bf*)(arow + k0 + half * 8);
        v8bf a1 = *(const v8bf*)(arow + k0 + 16 + half * 8);
        v8bf b0 = *(const v8bf*)(brow + k0 + half * 8);
        v8bf b1 = *(const v8bf*)(brow + k0 + 16 + half * 8);
        v16bf a = __builtin_shufflevector(a0, a1, 0, 1, 2, 3, 4, 5, 6, 7,
                                          8, 9, 10, 11, 12, 13, 14, 15);
        v16bf b = __builtin_shufflevector(b0, b1, 0, 1, 2, 3, 4, 5, 6, 7,
                                          8, 9, 10, 11, 12, 13, 14, 15);
        c = __builtin_amdgcn_wmma_f32_16x16x32_bf16(
                /*neg_a=*/false, a, /*neg_b=*/false, b,
                /*c_mod=*/(short)0, c, /*reuse_a=*/false, /*reuse_b=*/false);
    }

    // D layout: VGPR r, lane l -> M = m0 + r + 8*half, N = l16
    float bv = bias[(n < Nvalid) ? n : (Nvalid - 1)];
#pragma unroll
    for (int r = 0; r < 8; ++r) {
        int m = m0 + r + 8 * half;
        float v = fmaxf(c[r] + bv, 0.f);
        if (outb) {
            outb[(size_t)m * ldob + n] = (n < Nvalid) ? (__bf16)v : (__bf16)0.f;
        } else if (n < Nvalid) {
            outf[(size_t)m * ldof + n] = v;
        }
    }
}

// ---------------------------------------------------------------------------
// Head: fc3 (84->1) + RBF vs 10 supports + mean + sigmoid + concat.
// ---------------------------------------------------------------------------
__global__ void k_head(const float* __restrict__ h2,
                       const float* __restrict__ w3,
                       const float* __restrict__ b3,
                       const float* __restrict__ sup,
                       float* __restrict__ out)
{
    int b = blockIdx.x * blockDim.x + threadIdx.x;
    if (b >= 128) return;
    float s = b3[0];
#pragma unroll 4
    for (int k = 0; k < 84; ++k)
        s = fmaf(h2[b * 84 + k], w3[k], s);
    float acc = 0.f;
#pragma unroll
    for (int j = 0; j < 10; ++j) {
        float d = s - sup[j];
        acc += expf(-d * d);           // GAMMA = 1
    }
    float h = acc * 0.1f;              // mean over 10 supports
    float p = 1.f / (1.f + expf(-h));
    out[2 * b]     = p;
    out[2 * b + 1] = 1.f - p;
}

// ---------------------------------------------------------------------------
// Launch
// ---------------------------------------------------------------------------
extern "C" void kernel_launch(void* const* d_in, const int* in_sizes, int n_in,
                              void* d_out, int out_size, void* d_ws, size_t ws_size,
                              hipStream_t stream) {
    const float* x       = (const float*)d_in[0];   // [128,3,250,250]
    const float* conv1_w = (const float*)d_in[1];   // [6,3,5,5]
    const float* conv1_b = (const float*)d_in[2];   // [6]
    const float* conv2_w = (const float*)d_in[3];   // [15,6,3,3]
    const float* conv2_b = (const float*)d_in[4];   // [15]
    const float* fc1_w   = (const float*)d_in[5];   // [120,55815]
    const float* fc1_b   = (const float*)d_in[6];   // [120]
    const float* fc2_w   = (const float*)d_in[7];   // [84,120]
    const float* fc2_b   = (const float*)d_in[8];   // [84]
    const float* fc3_w   = (const float*)d_in[9];   // [1,84]
    const float* fc3_b   = (const float*)d_in[10];  // [1]
    const float* support = (const float*)d_in[11];  // [10,1]
    float* out = (float*)d_out;                     // [128,2]

    // Workspace layout (byte offsets, all 16B-aligned)
    char* ws = (char*)d_ws;
    size_t off = 0;
    float*  pool1 = (float*)(ws + off);  off += (size_t)128 * 6 * 123 * 123 * 4; // 46.5 MB
    __bf16* act   = (__bf16*)(ws + off); off += (size_t)128 * K1P * 2;           // 14.3 MB
    __bf16* wB1   = (__bf16*)(ws + off); off += (size_t)N1P * K1P * 2;           // 14.3 MB
    __bf16* h1b   = (__bf16*)(ws + off); off += (size_t)128 * K2P * 2;
    __bf16* wB2   = (__bf16*)(ws + off); off += (size_t)N2P * K2P * 2;
    float*  h2    = (float*)(ws + off);  off += (size_t)128 * 84 * 4;

    // Weight conversion / padding (deterministic, runs every launch)
    {
        size_t tot1 = (size_t)N1P * K1P;
        k_cvt_pad_bf16<<<dim3((unsigned)((tot1 + 255) / 256)), 256, 0, stream>>>(
            fc1_w, 120, K1, wB1, N1P, K1P);
        size_t tot2 = (size_t)N2P * K2P;
        k_cvt_pad_bf16<<<dim3((unsigned)((tot2 + 255) / 256)), 256, 0, stream>>>(
            fc2_w, 84, 120, wB2, N2P, K2P);
        k_zero_act_tail<<<dim3((128 * (K1P - K1) + 255) / 256), 256, 0, stream>>>(act);
    }

    // conv1 + relu + pool
    {
        dim3 grid((123 * 123 + 255) / 256, 6, 128);
        k_conv1pool<<<grid, 256, 0, stream>>>(x, conv1_w, conv1_b, pool1);
    }
    // conv2 + relu + pool -> bf16 activations [128, K1P]
    {
        dim3 grid((61 * 61 + 255) / 256, 15, 128);
        k_conv2pool<<<grid, 256, 0, stream>>>(pool1, conv2_w, conv2_b, act);
    }
    // fc1: WMMA bf16, branch-free K loop; bf16 padded output [128, K2P]
    {
        dim3 grid(128 / 16, N1P / 16);
        k_fc_wmma<<<grid, 32, 0, stream>>>(act, K1P, wB1, K1P, fc1_b,
                                           K1P, 120, h1b, K2P, nullptr, 0);
    }
    // fc2: WMMA bf16; fp32 output [128, 84]
    {
        dim3 grid(128 / 16, N2P / 16);
        k_fc_wmma<<<grid, 32, 0, stream>>>(h1b, K2P, wB2, K2P, fc2_b,
                                           K2P, 84, nullptr, 0, h2, 84);
    }
    // fc3 + RBF + mean + sigmoid + concat
    k_head<<<1, 128, 0, stream>>>(h2, fc3_w, fc3_b, support, out);
}
